// BaysianMLPMaskedDropout_20693152432476
// MI455X (gfx1250) — matrix operand order repaired and checked by reading, not compiled, not run on hardware
//
#include <hip/hip_runtime.h>
#include <hip/hip_bf16.h>
#include <cstdint>

// ---------------- problem constants ----------------
#define BATCH 1024
#define NIN   2048
#define NH0   4096
#define NH1   4096
#define NOUT  1000
#define EPS_CDF 1e-5f
#define DECAY   0.01f
#define NEG_SLOPE 0.01f

typedef __attribute__((ext_vector_type(16))) _Float16 v16h;
typedef __attribute__((ext_vector_type(8)))  _Float16 v8h;
typedef __attribute__((ext_vector_type(8)))  float    v8f;
typedef __attribute__((ext_vector_type(4)))  unsigned int u32x4;
typedef __attribute__((ext_vector_type(8)))  int          i32x8;
typedef __attribute__((ext_vector_type(4)))  int          i32x4;

// ---------------- helpers ----------------
__device__ __forceinline__ float wred(float v) {
#pragma unroll
  for (int m = 16; m > 0; m >>= 1) v += __shfl_xor(v, m, 32);
  return v;
}

__device__ __forceinline__ v16h ld_frag(const _Float16* p, int o0, int o1) {
  v8h a = *(const v8h*)(p + o0);
  v8h b = *(const v8h*)(p + o1);
  return __builtin_shufflevector(a, b, 0,1,2,3,4,5,6,7,8,9,10,11,12,13,14,15);
}

// ---------------- small kernels ----------------
__global__ void k_init_acc(float* acc) {
  if (threadIdx.x < 16) acc[threadIdx.x] = 0.0f;
}

// x -> f16 copy fused with Z = sum(x @ z_w)
__global__ void k_conv_x(const float* __restrict__ x, const float* __restrict__ zw,
                         _Float16* __restrict__ xh, size_t n, int inMask, float* accZ) {
  size_t i = (size_t)blockIdx.x * blockDim.x + threadIdx.x;
  float z = 0.0f;
  if (i < n) {
    float v = x[i];
    xh[i] = (_Float16)v;
    z = v * zw[(int)(i & (size_t)inMask)];
  }
  z = wred(z);
  if ((threadIdx.x & 31) == 0) atomicAdd(accZ, z);
}

// w = mu + exp(ls)*eps -> f16; fused log p_w and weight-decay reductions
__global__ void k_sample_w(const float* __restrict__ mu, const float* __restrict__ ls,
                           const float* __restrict__ eps, _Float16* __restrict__ wh,
                           size_t n, int doFlow, float* accFlow, float* accWd) {
  size_t i = (size_t)blockIdx.x * blockDim.x + threadIdx.x;
  float fl = 0.0f, wd = 0.0f;
  if (i < n) {
    float m = mu[i], s = __expf(ls[i]), e = eps[i];
    float w = m + s * e;
    wh[i] = (_Float16)w;
    if (doFlow) {
      // norm.cdf(w+eps,mu,s)-norm.cdf(w-eps,mu,s) = Phi(e+d)-Phi(e-d), d=EPS_CDF/s
      float d = EPS_CDF / s;
      float pw = 0.5f * (erff((e + d) * 0.70710678f) - erff((e - d) * 0.70710678f));
      fl = __logf(fmaxf(pw, 1e-38f));
      wd = DECAY * (m * m + s);
    }
  }
  fl = wred(fl); wd = wred(wd);
  if (((threadIdx.x & 31) == 0) && doFlow) { atomicAdd(accFlow, fl); atomicAdd(accWd, wd); }
}

// b = b_mu + exp(b_ls)*eps (kept f32); fused log p_b
__global__ void k_sample_b(const float* __restrict__ mu, const float* __restrict__ ls,
                           const float* __restrict__ eps, float* __restrict__ bs,
                           int n, int doFlow, float* accFlow) {
  int i = blockIdx.x * blockDim.x + threadIdx.x;
  float fl = 0.0f;
  if (i < n) {
    float m = mu[i], s = __expf(ls[i]), e = eps[i];
    bs[i] = m + s * e;
    if (doFlow) {
      float d = EPS_CDF / s;
      float pb = 0.5f * (erff((e + d) * 0.70710678f) - erff((e - d) * 0.70710678f));
      fl = __logf(fmaxf(pb, 1e-38f));
    }
  }
  fl = wred(fl);
  if (((threadIdx.x & 31) == 0) && doFlow) atomicAdd(accFlow, fl);
}

// per-row mask sum -> mult[b]; fused GFN flow of the z-mask
__global__ void k_mask_flow(const int* __restrict__ mz, const int* __restrict__ mask,
                            const float* __restrict__ pfz, float* __restrict__ mult,
                            int H, float* accFlow) {
  __shared__ float sS[256], sF[256];
  int b = blockIdx.x, tid = threadIdx.x;
  float ms = 0.0f, fl = 0.0f;
  for (int h = tid; h < H; h += 256) {
    int z = mz[(size_t)b * H + h];
    ms += (float)(z * mask[h]);
    float p = fminf(fmaxf(pfz[(size_t)b * H + h], 1e-6f), 1.0f - 1e-6f);
    fl += z ? __logf(p) : __logf(1.0f - p);
  }
  sS[tid] = ms; sF[tid] = fl; __syncthreads();
  for (int s = 128; s > 0; s >>= 1) {
    if (tid < s) { sS[tid] += sS[tid + s]; sF[tid] += sF[tid + s]; }
    __syncthreads();
  }
  if (tid == 0) {
    mult[b] = (float)H / (sS[0] + 1e-6f);
    atomicAdd(accFlow, sF[0]);
  }
}

// ---------------- TDM tile load: 2D slab global->LDS ----------------
// Loads tile [128 x 32] f16 from row-major [tensorRows x K] at gbase,
// inserting 16B LDS pad after every 64B row (stride 40 halfs). Rows beyond
// tensorRows are hardware zero-filled (OOB reads return 0).
#define TILE_M 128
#define TILE_N 128
#define TILE_K 32
#define LDSS   40   // 32 + 8 halfs pad (matches TDM pad_interval/amount)

__device__ __forceinline__ void tdm_load_tile(const _Float16* gbase, unsigned ldsAddr,
                                              int K, int tensorRows) {
  unsigned long long ga = (unsigned long long)(uintptr_t)gbase;
  u32x4 g0;
  g0[0] = 1u;                                             // count=1, user descriptor
  g0[1] = ldsAddr;                                        // LDS byte address
  g0[2] = (unsigned)(ga & 0xffffffffu);                   // global_addr[31:0]
  g0[3] = (unsigned)((ga >> 32) & 0x01ffffffu) | 0x80000000u; // [56:32] | type=2
  unsigned td0 = (unsigned)K;            // tensor_dim0 (halfs) >= tile width
  unsigned td1 = (unsigned)tensorRows;   // tensor_dim1: rows available (zero-fill past)
  unsigned long long st0 = (unsigned long long)K;  // row stride in halfs
  i32x8 g1;
  g1[0] = (int)((1u << 16)      // data_size = 1 (2 bytes)
              | (1u << 20)      // pad_enable
              | (3u << 22)      // pad_interval: 16 DWORDs (64B)
              | (3u << 25));    // pad_amount: 4 DWORDs (16B)
  g1[1] = (int)((td0 & 0xffffu) << 16);                   // tensor_dim0[15:0]
  g1[2] = (int)((td0 >> 16) | ((td1 & 0xffffu) << 16));   // dim0[31:16] | dim1[15:0]
  g1[3] = (int)((td1 >> 16) | ((unsigned)TILE_K << 16));  // dim1[31:16] | tile_dim0
  g1[4] = (int)(unsigned)TILE_M;                          // tile_dim1 | tile_dim2=0
  g1[5] = (int)(unsigned)(st0 & 0xffffffffu);             // dim0_stride[31:0]
  g1[6] = (int)(unsigned)((st0 >> 32) & 0xffffu);         // dim0_stride[47:32] | dim1_stride[15:0]=0
  g1[7] = 0;
  i32x4 z4 = {0, 0, 0, 0};
  i32x8 z8 = {0, 0, 0, 0, 0, 0, 0, 0};
  // 6-arg form (clang-23 / therock-10.0 toolchain)
  __builtin_amdgcn_tensor_load_to_lds(g0, g1, z4, z4, z8, 0);
}

// ---------------- WMMA GEMM: C[b,n] = sum_k A[b,k]*W[n,k] ----------------
// TDM double-buffered staging; 8 waves; 2x4 16x16 accum tiles per wave.
// EPI==0: h = leaky(acc+bias)*mz*mask*mult -> f16 (next-layer activation)
// EPI==1: pred = acc+bias -> f32 d_out (cols < Nout)
template <int EPI>
__global__ __launch_bounds__(256)
void k_gemm_wmma(const _Float16* __restrict__ A, const _Float16* __restrict__ W,
                 int M, int N, int K, int Nw,
                 const float* __restrict__ bias,
                 const int* __restrict__ mz, const int* __restrict__ mask_mu,
                 const float* __restrict__ mult,
                 _Float16* __restrict__ outH, float* __restrict__ outF, int Nout) {
  __shared__ _Float16 lA[2][TILE_M * LDSS];
  __shared__ _Float16 lB[2][TILE_N * LDSS];
  const int tid  = threadIdx.x;
  const int lane = tid & 31;
  const int waveId = __builtin_amdgcn_readfirstlane(tid) >> 5;  // scalar wave index
  const int wm = waveId >> 1;      // 0..3 -> 32-row strip
  const int wn = waveId & 1;       // 0..1 -> 64-col strip
  const int l16 = lane & 15;
  const int hs  = lane >> 4;       // half-wave select
  const int bm = blockIdx.y * TILE_M;
  const int bn = blockIdx.x * TILE_N;

  v8f acc[2][4] = {};

  // prologue: wave 0 kicks off DMA of K-slab 0 into buffer 0
  if (waveId == 0) {
    tdm_load_tile(A + (size_t)bm * K, (unsigned)(uintptr_t)&lA[0][0], K, M - bm);
    tdm_load_tile(W + (size_t)bn * K, (unsigned)(uintptr_t)&lB[0][0], K, Nw - bn);
  }

  int ibuf = 0;
  for (int k0 = 0; k0 < K; k0 += TILE_K, ibuf ^= 1) {
    const bool more = (k0 + TILE_K) < K;
    if (waveId == 0) {
      if (more) {
        // prefetch next K-slab into the other buffer (overlaps this slab's WMMAs)
        tdm_load_tile(A + (size_t)bm * K + (k0 + TILE_K),
                      (unsigned)(uintptr_t)&lA[ibuf ^ 1][0], K, M - bm);
        tdm_load_tile(W + (size_t)bn * K + (k0 + TILE_K),
                      (unsigned)(uintptr_t)&lB[ibuf ^ 1][0], K, Nw - bn);
        __builtin_amdgcn_s_wait_tensorcnt(2);  // current buffer's pair retired
      } else {
        __builtin_amdgcn_s_wait_tensorcnt(0);
      }
    }
    __syncthreads();  // current buffer visible to all waves

    v16h af[2];
#pragma unroll
    for (int mt = 0; mt < 2; mt++) {
      const _Float16* pa = &lA[ibuf][(wm * 32 + mt * 16 + l16) * LDSS];
      af[mt] = ld_frag(pa, hs * 8, hs * 8 + 16);     // A: K in 8-chunks
    }
#pragma unroll
    for (int nt = 0; nt < 4; nt++) {
      const _Float16* pb = &lB[ibuf][(wn * 64 + nt * 16 + l16) * LDSS];
      v16h bf = ld_frag(pb, hs * 8, hs * 8 + 16);    // B: K in 8-chunks
      acc[0][nt] = __builtin_amdgcn_wmma_f32_16x16x32_f16(
          false, af[0], false, bf, (short)0, acc[0][nt], false, false);
      acc[1][nt] = __builtin_amdgcn_wmma_f32_16x16x32_f16(
          false, af[1], false, bf, (short)0, acc[1][nt], false, false);
    }
    __syncthreads();  // all reads done before this buffer is re-targeted by DMA
  }

  // epilogue: C layout = VGPR r -> row r + 8*hs, col = lane%16
#pragma unroll
  for (int mt = 0; mt < 2; mt++) {
#pragma unroll
    for (int nt = 0; nt < 4; nt++) {
#pragma unroll
      for (int r = 0; r < 8; r++) {
        int gr = bm + wm * 32 + mt * 16 + r + hs * 8;
        int gc = bn + wn * 64 + nt * 16 + l16;
        float v = acc[mt][nt][r];
        if (EPI == 0) {
          v += bias[gc];
          v = (v >= 0.0f) ? v : NEG_SLOPE * v;
          float m = (float)(mz[(size_t)gr * N + gc] * mask_mu[gc]);
          v *= m * mult[gr];
          outH[(size_t)gr * N + gc] = (_Float16)v;
        } else {
          if (gc < Nout) {
            v += bias[gc];
            outF[(size_t)gr * Nout + gc] = v;
          }
        }
      }
    }
  }
}

// ---------------- cross entropy over pred rows ----------------
__global__ void k_ce(const float* __restrict__ pred, const int* __restrict__ y,
                     int C, float* accCE) {
  __shared__ float red[256];
  int b = blockIdx.x, tid = threadIdx.x;
  float mx = -1e30f;
  for (int c = tid; c < C; c += 256) mx = fmaxf(mx, pred[(size_t)b * C + c]);
  red[tid] = mx; __syncthreads();
  for (int s = 128; s > 0; s >>= 1) { if (tid < s) red[tid] = fmaxf(red[tid], red[tid + s]); __syncthreads(); }
  mx = red[0]; __syncthreads();
  float se = 0.0f;
  for (int c = tid; c < C; c += 256) se += __expf(pred[(size_t)b * C + c] - mx);
  red[tid] = se; __syncthreads();
  for (int s = 128; s > 0; s >>= 1) { if (tid < s) red[tid] += red[tid + s]; __syncthreads(); }
  if (tid == 0) {
    float lse = __logf(red[0]) + mx;
    float lp = pred[(size_t)b * C + y[b]] - lse;
    atomicAdd(accCE, -lp);
  }
}

__global__ void k_final(const float* __restrict__ acc, const int* __restrict__ dss,
                        float* __restrict__ outLoss) {
  float ds = (float)dss[0];
  float loss = acc[0] /*Z*/ + acc[1] /*flow*/ + ds * (acc[3] / (float)BATCH) /*NLL*/ + acc[2] /*wd*/;
  outLoss[0] = loss * loss;
}

// ---------------- launcher ----------------
extern "C" void kernel_launch(void* const* d_in, const int* in_sizes, int n_in,
                              void* d_out, int out_size, void* d_ws, size_t ws_size,
                              hipStream_t stream) {
  (void)in_sizes; (void)n_in; (void)out_size; (void)ws_size;
  const float* x      = (const float*)d_in[0];
  const float* zw     = (const float*)d_in[1];
  const float* w_mu0  = (const float*)d_in[2];
  const float* w_ls0  = (const float*)d_in[3];
  const float* b_mu0  = (const float*)d_in[4];
  const float* b_ls0  = (const float*)d_in[5];
  const float* eps_w0 = (const float*)d_in[6];
  const float* eps_b0 = (const float*)d_in[7];
  const float* w_mu1  = (const float*)d_in[8];
  const float* w_ls1  = (const float*)d_in[9];
  const float* b_mu1  = (const float*)d_in[10];
  const float* b_ls1  = (const float*)d_in[11];
  const float* eps_w1 = (const float*)d_in[12];
  const float* eps_b1 = (const float*)d_in[13];
  const float* w_muo  = (const float*)d_in[14];
  const float* w_lso  = (const float*)d_in[15];
  const float* b_muo  = (const float*)d_in[16];
  const float* b_lso  = (const float*)d_in[17];
  const float* eps_wo = (const float*)d_in[18];
  const float* eps_bo = (const float*)d_in[19];
  const float* pfz0   = (const float*)d_in[20];
  const float* pfz1   = (const float*)d_in[21];
  const int*   y      = (const int*)d_in[22];
  const int*   mz0    = (const int*)d_in[23];
  const int*   mz1    = (const int*)d_in[24];
  const int*   mmu0   = (const int*)d_in[25];
  const int*   mmu1   = (const int*)d_in[26];
  const int*   dss    = (const int*)d_in[27];

  float* pred = (float*)d_out;                       // [1024 x 1000]
  float* lossOut = pred + (size_t)BATCH * NOUT;      // scalar at the end

  // ---- workspace layout ----
  float* acc   = (float*)d_ws;           // [0]=Z [1]=flow [2]=wd [3]=ce_sum
  float* mult0 = acc + 16;               // [1024]
  float* mult1 = mult0 + BATCH;          // [1024]
  float* b0s   = mult1 + BATCH;          // [4096]
  float* b1s   = b0s + NH0;              // [4096]
  float* bos   = b1s + NH1;              // [1000] (pad 1024)
  _Float16* xh  = (_Float16*)(bos + 1024);
  _Float16* w0h = xh  + (size_t)BATCH * NIN;
  _Float16* w1h = w0h + (size_t)NH0 * NIN;
  _Float16* woh = w1h + (size_t)NH1 * NH0;
  _Float16* h0h = woh + (size_t)NOUT * NH1;
  _Float16* h1h = h0h + (size_t)BATCH * NH0;

  k_init_acc<<<1, 32, 0, stream>>>(acc);

  {
    size_t n = (size_t)BATCH * NIN;
    k_conv_x<<<(unsigned)((n + 255) / 256), 256, 0, stream>>>(x, zw, xh, n, NIN - 1, &acc[0]);
  }

  {
    size_t n0 = (size_t)NH0 * NIN;
    k_sample_w<<<(unsigned)((n0 + 255) / 256), 256, 0, stream>>>(w_mu0, w_ls0, eps_w0, w0h, n0, 1, &acc[1], &acc[2]);
    size_t n1 = (size_t)NH1 * NH0;
    k_sample_w<<<(unsigned)((n1 + 255) / 256), 256, 0, stream>>>(w_mu1, w_ls1, eps_w1, w1h, n1, 1, &acc[1], &acc[2]);
    size_t no = (size_t)NOUT * NH1;
    k_sample_w<<<(unsigned)((no + 255) / 256), 256, 0, stream>>>(w_muo, w_lso, eps_wo, woh, no, 0, &acc[1], &acc[2]);
  }

  k_sample_b<<<(NH0 + 255) / 256, 256, 0, stream>>>(b_mu0, b_ls0, eps_b0, b0s, NH0, 1, &acc[1]);
  k_sample_b<<<(NH1 + 255) / 256, 256, 0, stream>>>(b_mu1, b_ls1, eps_b1, b1s, NH1, 1, &acc[1]);
  k_sample_b<<<(NOUT + 255) / 256, 256, 0, stream>>>(b_muo, b_lso, eps_bo, bos, NOUT, 0, &acc[1]);

  k_mask_flow<<<BATCH, 256, 0, stream>>>(mz0, mmu0, pfz0, mult0, NH0, &acc[1]);
  k_mask_flow<<<BATCH, 256, 0, stream>>>(mz1, mmu1, pfz1, mult1, NH1, &acc[1]);

  // layer 0: [1024,2048] x [4096,2048]^T -> h0
  k_gemm_wmma<0><<<dim3(NH0 / TILE_N, BATCH / TILE_M), 256, 0, stream>>>(
      xh, w0h, BATCH, NH0, NIN, NH0, b0s, mz0, mmu0, mult0, h0h, nullptr, 0);
  // layer 1: [1024,4096] x [4096,4096]^T -> h1
  k_gemm_wmma<0><<<dim3(NH1 / TILE_N, BATCH / TILE_M), 256, 0, stream>>>(
      h0h, w1h, BATCH, NH1, NH0, NH1, b1s, mz1, mmu1, mult1, h1h, nullptr, 0);
  // output: [1024,4096] x [1000,4096]^T -> pred (N padded to 1024, TDM zero-fills rows >= 1000)
  k_gemm_wmma<1><<<dim3(1024 / TILE_N, BATCH / TILE_M), 256, 0, stream>>>(
      h1h, woh, BATCH, 1024, NH1, NOUT, bos, nullptr, nullptr, nullptr, nullptr, pred, NOUT);

  k_ce<<<BATCH, 256, 0, stream>>>(pred, y, NOUT, &acc[3]);
  k_final<<<1, 1, 0, stream>>>(acc, dss, lossOut);
}
